// RWKV_x_71665824301908
// MI455X (gfx1250) — compile-verified, matrix operand order checked by reading, not run on hardware
//
#include <hip/hip_runtime.h>
#include <hip/hip_bf16.h>

// ---------------------------------------------------------------------------
// Problem shapes (fixed by the reference)
// ---------------------------------------------------------------------------
#define Bsz  4
#define Tlen 512
#define Hh   32
#define Nn   64
#define Dd   2048   // = Hh * Nn, also GEMM M (=B*T), N, K

#define BK   64            // GEMM K per pipeline stage (2 WMMA-K chunks)
#define PADH 8             // LDS row padding (halves): 72-half rows -> the
#define AROW (BK + PADH)   // 16-lane ds_load_b128 pattern hits 16 banks

typedef __attribute__((ext_vector_type(16))) _Float16 v16h;
typedef __attribute__((ext_vector_type(8)))  _Float16 h8;
typedef __attribute__((ext_vector_type(8)))  float    v8f;

// ---------------------------------------------------------------------------
// Kernel 1: RWKV x070 WKV scan.
// One block per (b,h). 256 threads: thread = (row i in 0..63, part in 0..3),
// each thread owns state[i][16*part .. 16*part+15] in registers.
// Next timestep's inputs are prefetched into registers before the barrier so
// global-load latency overlaps the current step's compute (scan is
// latency-bound: 512 serial steps).
// ---------------------------------------------------------------------------
__global__ __launch_bounds__(256) void wkv7_kernel(
    const float* __restrict__ r, const float* __restrict__ w,
    const float* __restrict__ k, const float* __restrict__ v,
    const float* __restrict__ iclr,
    _Float16* __restrict__ yh /* [B*T, D] f16, A-matrix for the GEMM */) {

  const int bh   = blockIdx.x;       // 0..127
  const int b    = bh / Hh;
  const int h    = bh % Hh;
  const int tid  = threadIdx.x;
  const int i    = tid & 63;         // state row
  const int part = tid >> 6;         // 0..3, owns j in [16*part, 16*part+16)
  const int j0   = part * 16;

  __shared__ float r_s[Nn], d_s[Nn], k_s[Nn], v_s[Nn], a_s[Nn], b_s[Nn];
  __shared__ float part_sa[4][Nn];
  __shared__ float part_y[4][Nn];

  float st[16];
#pragma unroll
  for (int jj = 0; jj < 16; ++jj) st[jj] = 0.0f;

  const size_t strideT = (size_t)Hh * Nn;
  size_t base = (size_t)b * Tlen * strideT + (size_t)h * Nn;

  // register prefetch of step-t inputs (part==0 threads only)
  float ld_r = 0.f, ld_w = 0.f, ld_k = 0.f, ld_v = 0.f, ld_ic = 0.f;
  if (part == 0) {
    ld_r  = r[base + i];  ld_w = w[base + i];  ld_k = k[base + i];
    ld_v  = v[base + i];  ld_ic = iclr[base + i];
  }

  for (int t = 0; t < Tlen; ++t) {
    if (part == 0) {
      k_s[i] = ld_k;
      r_s[i] = ld_r;
      v_s[i] = ld_v;
      d_s[i] = __expf(-__expf(ld_w));                 // decay in (0,1)
      b_s[i] = 1.0f / (1.0f + __expf(-ld_ic));        // sigmoid (temp)
      if (t + 1 < Tlen) {                             // prefetch next step
        size_t nb = base + strideT + i;
        ld_r = r[nb]; ld_w = w[nb]; ld_k = k[nb]; ld_v = v[nb]; ld_ic = iclr[nb];
      }
    }
    __syncthreads();

    // ||k||^2 via broadcast LDS reads (same addr across lanes -> broadcast)
    float ss = 0.0f;
#pragma unroll 16
    for (int j = 0; j < Nn; ++j) ss += k_s[j] * k_s[j];
    float rinv = rsqrtf(ss + 1e-12f);

    if (part == 0) {
      float kk = k_s[i] * rinv;
      a_s[i] = -kk;                  // removal direction
      b_s[i] = kk * b_s[i];          // kk * sigmoid(iclr)
    }
    __syncthreads();

    // sa[i] = sum_j state[i][j] * a[j]   (4-way partial reduction)
    float p = 0.0f;
#pragma unroll
    for (int jj = 0; jj < 16; ++jj) p += st[jj] * a_s[j0 + jj];
    part_sa[part][i] = p;
    __syncthreads();
    float sa = part_sa[0][i] + part_sa[1][i] + part_sa[2][i] + part_sa[3][i];

    // state update + y partial
    const float vv = v_s[i];
    float yp = 0.0f;
#pragma unroll
    for (int jj = 0; jj < 16; ++jj) {
      const int j = j0 + jj;
      float s = st[jj] * d_s[j] + sa * b_s[j] + vv * k_s[j];
      st[jj] = s;
      yp += s * r_s[j];
    }
    part_y[part][i] = yp;
    __syncthreads();

    if (part == 0) {
      float y = part_y[0][i] + part_y[1][i] + part_y[2][i] + part_y[3][i];
      yh[((size_t)b * Tlen + t) * Dd + h * Nn + i] = (_Float16)y;
    }
    base += strideT;
    __syncthreads();  // protect shared vectors before next step's writes
  }
}

// ---------------------------------------------------------------------------
// Kernel 2: int32 quantized weights -> f16 (values in [-128,127] are exact).
// Per-row scale is applied in the GEMM epilogue in f32.
// ---------------------------------------------------------------------------
__global__ __launch_bounds__(256) void quantcvt_kernel(
    const int* __restrict__ wq, _Float16* __restrict__ wh, int n) {
  int idx = blockIdx.x * blockDim.x + threadIdx.x;
  if (idx < n) wh[idx] = (_Float16)wq[idx];
}

// ---------------------------------------------------------------------------
// CDNA5 async global->LDS copy helper (ASYNCcnt-tracked; gfx1250 path).
// The generic pointer to a __shared__ object carries the LDS byte offset in
// its low 32 bits.
// ---------------------------------------------------------------------------
__device__ __forceinline__ void async_copy_b128(const void* g, void* lds) {
  uint32_t l = (uint32_t)(uintptr_t)lds;
  asm volatile("global_load_async_to_lds_b128 %0, %1, off"
               :: "v"(l), "v"(g) : "memory");
}

// WMMA fragment from an LDS tile row (row-major, AROW-half row stride).
// CDNA5 16-bit A/B layout: lane L -> row L%16, K chunks [kb,kb+8) and
// [kb+16,kb+24) with kb=(L/16)*8 -> two 16-byte ds_load_b128 per fragment.
__device__ __forceinline__ v16h frag_from_lds(const _Float16* p) {
  h8 lo = *reinterpret_cast<const h8*>(p);        // k .. k+7
  h8 hi = *reinterpret_cast<const h8*>(p + 16);   // k+16 .. k+23
  v16h f;
#pragma unroll
  for (int e = 0; e < 8; ++e) { f[e] = lo[e]; f[8 + e] = hi[e]; }
  return f;
}

// ---------------------------------------------------------------------------
// Kernel 3: out[m, n] = scale[n] * sum_k A[m,k] * W[n,k]
// M = N = K = 2048. Block = 256 threads = 8 waves.
// Block tile 64(M) x 128(N); waves arranged 4(M) x 2(N), wave tile 16 x 64
// with 4 in-place accumulators. K pipeline: double-buffered LDS tiles
// (BK=64, rows padded to 72 halves) filled by async global->LDS copies.
// Every thread issues exactly 6 b128 async copies per stage, so
// "s_wait_asynccnt 0x6" retires exactly the older stage while the newer one
// stays in flight; 8 WMMAs execute between each barrier pair.
// ---------------------------------------------------------------------------
__global__ __launch_bounds__(256) void gemm_f16_wmma_kernel(
    const _Float16* __restrict__ A,    // [2048, 2048] row-major (y in f16)
    const _Float16* __restrict__ Wt,   // [2048, 2048] row-major: row n = out ch
    const float* __restrict__ scale,   // [2048] per-output-row dequant scale
    float* __restrict__ out) {         // [2048, 2048] f32

  __shared__ _Float16 Asm[2][64][AROW];    // 2 x  9 KB
  __shared__ _Float16 Bsm[2][128][AROW];   // 2 x 18 KB   (total ~55 KB)

  const int tid  = threadIdx.x;
  const int lane = tid & 31;
  const int wv   = tid >> 5;                   // 0..7
  const int m0   = blockIdx.x * 64;            // block M base
  const int n0b  = blockIdx.y * 128;           // block N base
  const int mt4  = wv >> 1;                    // wave: which 16-row M sub-tile
  const int nh   = wv & 1;                     // wave: which 64-col N half

  // --- async stage loader: every thread issues exactly 6 b128 copies ---
  const int rTh = tid >> 3;                    // 0..31 (tile row slice)
  const int c8  = (tid & 7) * 8;               // col chunk (halves), 16B

  auto load_stage = [&](int buf, int k0) {
#pragma unroll
    for (int qa = 0; qa < 2; ++qa) {           // A: 64 rows
      const int rA = qa * 32 + rTh;
      async_copy_b128(A + (size_t)(m0 + rA) * Dd + k0 + c8,
                      &Asm[buf][rA][c8]);
    }
#pragma unroll
    for (int q = 0; q < 4; ++q) {              // B: 128 rows
      const int rB = q * 32 + rTh;
      async_copy_b128(Wt + (size_t)(n0b + rB) * Dd + k0 + c8,
                      &Bsm[buf][rB][c8]);
    }
  };

  v8f acc[4];
#pragma unroll
  for (int f = 0; f < 4; ++f) acc[f] = (v8f){};

  const int row = lane & 15;
  const int kb  = (lane >> 4) * 8;
  const int NK  = Dd / BK;                     // 32 pipeline stages

  load_stage(0, 0);                            // prologue: fill stage 0

  for (int ks = 0; ks < NK; ++ks) {
    const int cur = ks & 1;
    if (ks + 1 < NK) {
      load_stage(cur ^ 1, (ks + 1) * BK);      // overlap next stage's copies
      asm volatile("s_wait_asynccnt 0x6" ::: "memory");  // older stage done
    } else {
      asm volatile("s_wait_asynccnt 0x0" ::: "memory");
    }
    __syncthreads();                           // stage 'cur' visible to all

#pragma unroll
    for (int kc = 0; kc < 2; ++kc) {           // two WMMA-K chunks per stage
      const int kcol = kc * 32 + kb;
      v16h a = frag_from_lds(&Asm[cur][mt4 * 16 + row][kcol]);
      v16h bfr[4];
#pragma unroll
      for (int f = 0; f < 4; ++f)
        bfr[f] = frag_from_lds(&Bsm[cur][nh * 64 + f * 16 + row][kcol]);
#pragma unroll
      for (int f = 0; f < 4; ++f)
        acc[f] = __builtin_amdgcn_wmma_f32_16x16x32_f16(
            false, a, false, bfr[f], (short)0, acc[f], false, false);
    }

    __syncthreads();                           // all reads done before reuse
  }

  // Epilogue: C/D f32 layout: n = lane%16; VGPR e -> m = e + (lane/16)*8.
  const int mrow = m0 + mt4 * 16 + (lane >> 4) * 8;
#pragma unroll
  for (int f = 0; f < 4; ++f) {
    const int n = n0b + nh * 64 + f * 16 + (lane & 15);
    const float sc = scale[n];
#pragma unroll
    for (int e = 0; e < 8; ++e)
      out[(size_t)(mrow + e) * Dd + n] = acc[f][e] * sc;
  }
}

// ---------------------------------------------------------------------------
// Host-side launcher
// ---------------------------------------------------------------------------
extern "C" void kernel_launch(void* const* d_in, const int* in_sizes, int n_in,
                              void* d_out, int out_size, void* d_ws, size_t ws_size,
                              hipStream_t stream) {
  const float* r     = (const float*)d_in[0];
  const float* w     = (const float*)d_in[1];
  const float* k     = (const float*)d_in[2];
  const float* v     = (const float*)d_in[3];
  const float* iclr  = (const float*)d_in[4];
  const int*   wq    = (const int*)d_in[5];
  const float* wsc   = (const float*)d_in[6];
  float*       out   = (float*)d_out;

  // Workspace layout: [ y_f16 : 2048*2048 ][ w_f16 : 2048*2048 ]  = 16 MB
  _Float16* yh = (_Float16*)d_ws;
  _Float16* wh = yh + (size_t)Dd * Dd;

  // 1) WKV scan -> y (f16), one block per (b,h)
  wkv7_kernel<<<Bsz * Hh, 256, 0, stream>>>(r, w, k, v, iclr, yh);

  // 2) int32 weights -> f16 (exact for [-128,127])
  const int nW = Dd * Dd;
  quantcvt_kernel<<<nW / 256, 256, 0, stream>>>(wq, wh, nW);

  // 3) WMMA GEMM with async-LDS double-buffered pipeline + dequant epilogue
  dim3 grid(Dd / 64, Dd / 128);  // 32 x 16
  gemm_f16_wmma_kernel<<<grid, 256, 0, stream>>>(yh, wh, wsc, out);
}
